// ScoreNet_43843026158326
// MI455X (gfx1250) — matrix operand order repaired and checked by reading, not compiled, not run on hardware
//
#include <hip/hip_runtime.h>

#define B_ 16
#define L_ 2048
#define C_ 256
#define N_ 1024       // folded sequence length
#define HR_ 192       // H*R
#define HEAD_SCALE 0.17677669529663687f  // 1/sqrt(32)

typedef __attribute__((ext_vector_type(16))) __bf16        bf16x16;
typedef __attribute__((ext_vector_type(8)))  float         f32x8;
typedef __attribute__((ext_vector_type(4)))  unsigned int  u32x4;

union Frag {
    bf16x16 v;
    u32x4   u[2];   // 32 bytes = two b128 loads
};

// A fragment (16x32 bf16): lane<16 -> K 0..7 (V0-3) + 16..23 (V4-7);
// lane>=16 -> K 8..15 + 24..31.  p already includes (lane>>4)*8.
__device__ __forceinline__ void load_a_frag(const __bf16* p, Frag& f) {
    f.u[0] = *(const u32x4*)p;
    f.u[1] = *(const u32x4*)(p + 16);
}
// B fragment (32x16 bf16): lane<16 -> col lane, K 0..7 + 16..23; lane>=16 -> K 8..15 + 24..31.
// p already includes (lane>>4)*8.
__device__ __forceinline__ void load_b_frag(const __bf16* p, Frag& f) {
    f.u[0] = *(const u32x4*)p;
    f.u[1] = *(const u32x4*)(p + 16);
}

#define WMMA_BF16(A, Bm, Cacc) \
    __builtin_amdgcn_wmma_f32_16x16x32_bf16(false, (A), false, (Bm), (short)0, (Cacc), false, false)

// ---------------------------------------------------------------------------
// Pass 1: fold vertex pairs -> x (bf16), and d = x @ Wd^T + bd (f32)
// ---------------------------------------------------------------------------
__global__ __launch_bounds__(256) void fold_kernel(
    const float* __restrict__ feats, const float* __restrict__ Wd,
    const float* __restrict__ bd,
    __bf16* __restrict__ xbf, float* __restrict__ dbuf)
{
    const int n = blockIdx.x;
    const int b = blockIdx.y;
    const int c = threadIdx.x;
    const size_t base = ((size_t)b * (L_ + 1) + 1 + 2 * n) * C_ + c;
    const float xv = 0.5f * (feats[base] + feats[base + C_]);
    xbf[((size_t)b * N_ + n) * C_ + c] = (__bf16)xv;

    float p = xv * Wd[c];
#pragma unroll
    for (int off = 16; off; off >>= 1) p += __shfl_xor(p, off, 32);
    __shared__ float red[8];
    if ((threadIdx.x & 31) == 0) red[threadIdx.x >> 5] = p;
    __syncthreads();
    if (threadIdx.x == 0) {
        float s = 0.f;
#pragma unroll
        for (int i = 0; i < 8; ++i) s += red[i];
        dbuf[(size_t)b * N_ + n] = s + bd[0];
    }
}

// ---------------------------------------------------------------------------
// Tiny pass: convert Wq/Wk (f32 192x256 row-major) to bf16
// ---------------------------------------------------------------------------
__global__ __launch_bounds__(256) void wcvt_kernel(
    const float* __restrict__ Wq, const float* __restrict__ Wk,
    __bf16* __restrict__ wqbf, __bf16* __restrict__ wkbf)
{
    const int row = blockIdx.x;
    const int mat = blockIdx.y;
    const int c   = threadIdx.x;
    const float* src = mat ? Wk : Wq;
    __bf16*      dst = mat ? wkbf : wqbf;
    dst[(size_t)row * C_ + c] = (__bf16)src[(size_t)row * C_ + c];
}

// ---------------------------------------------------------------------------
// Pass 2: Q = x @ Wq^T, K = x @ Wk^T  (bf16 in/out, f32 accumulate)
// One wave per 16-row band; wave computes 16 x 192; K-loop over 256.
// Software-pipelined: next A/B fragments load while current WMMA executes.
// ---------------------------------------------------------------------------
__global__ __launch_bounds__(32) void proj_kernel(
    const __bf16* __restrict__ xbf,
    const __bf16* __restrict__ wqbf, const __bf16* __restrict__ wkbf,
    __bf16* __restrict__ qbuf, __bf16* __restrict__ kbuf)
{
    const int lane  = threadIdx.x;
    const int rows0 = blockIdx.x * 16;
    const int mat   = blockIdx.y;
    const __bf16* W   = mat ? wkbf : wqbf;
    __bf16*       Out = mat ? kbuf : qbuf;

    f32x8 acc[12];
#pragma unroll
    for (int i = 0; i < 12; ++i) acc[i] = (f32x8)0.f;

    const int aRow = rows0 + (lane & 15);
    const int aOff = (lane >> 4) * 8;
    const int bRow = lane & 15;
    const int bOff = (lane >> 4) * 8;

    const __bf16* aBase = xbf + (size_t)aRow * C_ + aOff;

    Frag a[2], bf_[2];
    load_a_frag(aBase, a[0]);
    load_b_frag(W + (size_t)bRow * C_ + bOff, bf_[0]);

#pragma unroll
    for (int ks = 0; ks < 8; ++ks) {          // K = 256 = 8 * 32
        const int kb  = ks * 32;
        const int cA  = ks & 1;
        if (ks < 7) load_a_frag(aBase + kb + 32, a[cA ^ 1]);
#pragma unroll
        for (int nt = 0; nt < 12; ++nt) {     // N = 192 = 12 * 16
            const int cB = (ks * 12 + nt) & 1;
            // prefetch next B fragment (next nt, or nt=0 of next k-step)
            if (nt < 11) {
                load_b_frag(W + (size_t)((nt + 1) * 16 + bRow) * C_ + kb + bOff,
                            bf_[cB ^ 1]);
            } else if (ks < 7) {
                load_b_frag(W + (size_t)bRow * C_ + (kb + 32) + bOff,
                            bf_[cB ^ 1]);
            }
            acc[nt] = WMMA_BF16(a[cA].v, bf_[cB].v, acc[nt]);
        }
    }

    const int rOff  = (lane >> 4) * 8;
    const int cLane = lane & 15;
#pragma unroll
    for (int nt = 0; nt < 12; ++nt) {
#pragma unroll
        for (int v = 0; v < 8; ++v) {
            const int r = rows0 + v + rOff;
            Out[(size_t)r * HR_ + nt * 16 + cLane] = (__bf16)acc[nt][v];
        }
    }
}

// ---------------------------------------------------------------------------
// Pass 3: S[b] = (Q[b] @ K[b]^T) * HEAD_SCALE (+ diag d) * max(ls, 0.01)
// WG = 256 threads -> 128x128 tile; 8 waves in 4(M) x 2(N), wave = 32x64.
// Double-buffered k-loop: fragments for step k+1 load during step k's WMMAs.
// S streamed out with non-temporal stores (write-BW bound: 64 MiB).
// ---------------------------------------------------------------------------
__global__ __launch_bounds__(256) void score_kernel(
    const __bf16* __restrict__ qbuf, const __bf16* __restrict__ kbuf,
    const float* __restrict__ dbuf, const float* __restrict__ lsp,
    float* __restrict__ out)
{
    const int lane  = threadIdx.x & 31;
    const int wid   = threadIdx.x >> 5;
    const int waveN = wid & 1;
    const int waveM = wid >> 1;
    const int b     = blockIdx.z;
    const int tileM = blockIdx.y * 128;
    const int tileN = blockIdx.x * 128;

    const __bf16* Qb = qbuf + (size_t)b * N_ * HR_;
    const __bf16* Kb = kbuf + (size_t)b * N_ * HR_;

    f32x8 acc[8];
#pragma unroll
    for (int i = 0; i < 8; ++i) acc[i] = (f32x8)0.f;

    const int aRow = tileM + waveM * 32 + (lane & 15);
    const int aOff = (lane >> 4) * 8;
    const int bRow = tileN + waveN * 64 + (lane & 15);
    const int bOff = (lane >> 4) * 8;

    const __bf16* aBase0 = Qb + (size_t)aRow * HR_ + aOff;
    const __bf16* aBase1 = Qb + (size_t)(aRow + 16) * HR_ + aOff;
    const __bf16* bBase0 = Kb + (size_t)bRow * HR_ + bOff;
    const __bf16* bBase1 = Kb + (size_t)(bRow + 16) * HR_ + bOff;
    const __bf16* bBase2 = Kb + (size_t)(bRow + 32) * HR_ + bOff;
    const __bf16* bBase3 = Kb + (size_t)(bRow + 48) * HR_ + bOff;

    Frag a[2][2], bb[2][4];
    load_a_frag(aBase0, a[0][0]);
    load_a_frag(aBase1, a[0][1]);
    load_b_frag(bBase0, bb[0][0]);
    load_b_frag(bBase1, bb[0][1]);
    load_b_frag(bBase2, bb[0][2]);
    load_b_frag(bBase3, bb[0][3]);

#pragma unroll
    for (int ks = 0; ks < 6; ++ks) {      // K = 192 = 6 * 32
        const int cur = ks & 1;
        const int nxt = cur ^ 1;
        if (ks < 5) {
            const int kb = (ks + 1) * 32;
            load_a_frag(aBase0 + kb, a[nxt][0]);
            load_a_frag(aBase1 + kb, a[nxt][1]);
            load_b_frag(bBase0 + kb, bb[nxt][0]);
            load_b_frag(bBase1 + kb, bb[nxt][1]);
            load_b_frag(bBase2 + kb, bb[nxt][2]);
            load_b_frag(bBase3 + kb, bb[nxt][3]);
        }
#pragma unroll
        for (int mt = 0; mt < 2; ++mt)
#pragma unroll
            for (int nt = 0; nt < 4; ++nt)
                acc[mt * 4 + nt] =
                    WMMA_BF16(a[cur][mt].v, bb[cur][nt].v, acc[mt * 4 + nt]);
    }

    const float ls    = fmaxf(lsp[0], 0.01f);
    const int   rOff  = (lane >> 4) * 8;
    const int   cLane = lane & 15;
#pragma unroll
    for (int mt = 0; mt < 2; ++mt) {
#pragma unroll
        for (int nt = 0; nt < 4; ++nt) {
            const int r0 = tileM + waveM * 32 + mt * 16 + rOff;
            const int c0 = tileN + waveN * 64 + nt * 16 + cLane;
#pragma unroll
            for (int v = 0; v < 8; ++v) {
                const int r = r0 + v;
                float val = acc[mt * 4 + nt][v] * HEAD_SCALE;
                if (r == c0) val += dbuf[(size_t)b * N_ + r];
                val *= ls;
                __builtin_nontemporal_store(
                    val, out + (size_t)b * N_ * N_ + (size_t)r * N_ + c0);
            }
        }
    }
}

// ---------------------------------------------------------------------------
extern "C" void kernel_launch(void* const* d_in, const int* in_sizes, int n_in,
                              void* d_out, int out_size, void* d_ws, size_t ws_size,
                              hipStream_t stream)
{
    const float* feats = (const float*)d_in[0];
    const float* Wq    = (const float*)d_in[1];
    const float* Wk    = (const float*)d_in[2];
    const float* Wd    = (const float*)d_in[3];
    const float* bd    = (const float*)d_in[4];
    const float* ls    = (const float*)d_in[5];
    float*       out   = (float*)d_out;

    char* ws = (char*)d_ws;
    size_t off = 0;
    __bf16* xbf  = (__bf16*)(ws + off); off += (size_t)B_ * N_ * C_ * 2;   // 8 MiB
    __bf16* qbuf = (__bf16*)(ws + off); off += (size_t)B_ * N_ * HR_ * 2;  // 6 MiB
    __bf16* kbuf = (__bf16*)(ws + off); off += (size_t)B_ * N_ * HR_ * 2;  // 6 MiB
    __bf16* wqbf = (__bf16*)(ws + off); off += (size_t)HR_ * C_ * 2;
    __bf16* wkbf = (__bf16*)(ws + off); off += (size_t)HR_ * C_ * 2;
    float*  dbuf = (float*)(ws + off);  off += (size_t)B_ * N_ * 4;
    if (off > ws_size) return;  // workspace too small (~20.3 MiB needed)

    fold_kernel <<<dim3(N_, B_),            256, 0, stream>>>(feats, Wd, bd, xbf, dbuf);
    wcvt_kernel <<<dim3(HR_, 2),            256, 0, stream>>>(Wq, Wk, wqbf, wkbf);
    proj_kernel <<<dim3((B_ * N_) / 16, 2),  32, 0, stream>>>(xbf, wqbf, wkbf, qbuf, kbuf);
    score_kernel<<<dim3(N_ / 128, N_ / 128, B_), 256, 0, stream>>>(qbuf, kbuf, dbuf, ls, out);
}
